// ODERNN_46849503265337
// MI455X (gfx1250) — matrix operand order repaired and checked by reading, not compiled, not run on hardware
//
#include <hip/hip_runtime.h>
#include <math.h>

// ---------------- problem constants ----------------
#define BATCH   1024
#define TSTEPS  128
#define XDIM    64
#define ZDIM    512
#define HDIM    512
#define MT      32          // batch rows per workgroup
#define LDD     96          // padded K for the GRU input GEMM (X+1=65 -> 96)
#define NTHREADS 256        // 8 wave32 waves

typedef __attribute__((ext_vector_type(16))) __bf16 bf16x16;
typedef __attribute__((ext_vector_type(8)))  float  f32x8;

// ---------------- fast activations (hardware v_tanh_f32 on gfx1250) ----------------
__device__ __forceinline__ float fast_tanh(float x) {
#if __has_builtin(__builtin_amdgcn_tanhf)
    return __builtin_amdgcn_tanhf(x);
#else
    float e = __builtin_amdgcn_exp2f(x * 2.8853900817779268f);
    return 1.f - 2.f * __builtin_amdgcn_rcpf(1.f + e);
#endif
}
__device__ __forceinline__ float fast_sigmoid(float x) {
    return 0.5f * fast_tanh(0.5f * x) + 0.5f;
}

// ---------------- helpers ----------------
__device__ __forceinline__ f32x8 bias_acc(float bv) {
    f32x8 a;
#pragma unroll
    for (int j = 0; j < 8; ++j) a[j] = bv;
    return a;
}

#define WMMA_BF16(af, bf, acc) \
    __builtin_amdgcn_wmma_f32_16x16x32_bf16(false, (af), false, (bf), (short)0, (acc), false, false)

// A-frag: lane (M = lane&15): halves {k0+16h..+15} contiguous          (LDS, row-major)
// B-frag: lane (N = lane&15): halves {k0+16h..+15} contiguous          (global, torch [out][in])
__device__ __forceinline__ bf16x16 load_a(const __bf16* pa, int k0) {
    bf16x16 af;
    __builtin_memcpy(&af, pa + k0, 32);                   // 2x ds_load_b128
    return af;
}
__device__ __forceinline__ bf16x16 load_b(const __bf16* pb, int k0) {
    bf16x16 bf;
    __builtin_memcpy(&bf, pb + k0, 32);                   // 2x global_load_b128
    return bf;
}

// Four adjacent 16-col tiles (ncol0 .. ncol0+63), shared A fragments, 4 indep. chains.
__device__ __forceinline__ void wmma_tile4(const __bf16* A, int lda, int mrow0,
                                           const __bf16* W, int ldw, int ncol0, int K,
                                           f32x8& a0, f32x8& a1, f32x8& a2, f32x8& a3,
                                           int nl, int half) {
    const __bf16* pa  = A + (size_t)(mrow0 + nl) * lda + half * 16;
    const __bf16* pb0 = W + (size_t)(ncol0 + nl) * ldw + half * 16;
    const __bf16* pb1 = pb0 + (size_t)16 * ldw;
    const __bf16* pb2 = pb0 + (size_t)32 * ldw;
    const __bf16* pb3 = pb0 + (size_t)48 * ldw;
#pragma unroll 2
    for (int k0 = 0; k0 < K; k0 += 32) {
        bf16x16 af = load_a(pa, k0);
        bf16x16 b0 = load_b(pb0, k0);
        bf16x16 b1 = load_b(pb1, k0);
        bf16x16 b2 = load_b(pb2, k0);
        bf16x16 b3 = load_b(pb3, k0);
        a0 = WMMA_BF16(af, b0, a0);
        a1 = WMMA_BF16(af, b1, a1);
        a2 = WMMA_BF16(af, b2, a2);
        a3 = WMMA_BF16(af, b3, a3);
    }
}

// Three tiles at column offsets {ncol, Z+ncol, 2Z+ncol} (GRU r/z/n gate sections),
// shared A fragments, 3 independent wmma chains.
__device__ __forceinline__ void wmma_tile3(const __bf16* A, int lda, int mrow0,
                                           const __bf16* W, int ldw, int ncol, int K,
                                           f32x8& a0, f32x8& a1, f32x8& a2,
                                           int nl, int half) {
    const __bf16* pa  = A + (size_t)(mrow0 + nl) * lda + half * 16;
    const __bf16* pb0 = W + (size_t)(ncol + nl) * ldw + half * 16;
    const __bf16* pb1 = pb0 + (size_t)ZDIM * ldw;
    const __bf16* pb2 = pb1 + (size_t)ZDIM * ldw;
#pragma unroll 2
    for (int k0 = 0; k0 < K; k0 += 32) {
        bf16x16 af = load_a(pa, k0);
        bf16x16 b0 = load_b(pb0, k0);
        bf16x16 b1 = load_b(pb1, k0);
        bf16x16 b2 = load_b(pb2, k0);
        a0 = WMMA_BF16(af, b0, a0);
        a1 = WMMA_BF16(af, b1, a1);
        a2 = WMMA_BF16(af, b2, a2);
    }
}

// Single tile (small output head)
__device__ __forceinline__ f32x8 wmma_tile1(const __bf16* A, int lda, int mrow0,
                                            const __bf16* W, int ldw, int ncol0, int K,
                                            f32x8 acc, int nl, int half) {
    const __bf16* pa = A + (size_t)(mrow0 + nl) * lda + half * 16;
    const __bf16* pb = W + (size_t)(ncol0 + nl) * ldw + half * 16;
#pragma unroll 4
    for (int k0 = 0; k0 < K; k0 += 32) {
        bf16x16 af = load_a(pa, k0);
        bf16x16 bf = load_b(pb, k0);
        acc = WMMA_BF16(af, bf, acc);
    }
    return acc;
}

__device__ __forceinline__ void build_data(__bf16* D, const float* x, const int* mask,
                                           int m0, int t) {
    for (int i = threadIdx.x; i < MT * LDD; i += NTHREADS) {
        int r = i / LDD, c = i - r * LDD;
        int b = m0 + r;
        float mk = (float)mask[(size_t)b * TSTEPS + t];
        float v = 0.f;
        if (c < XDIM)       v = x[((size_t)b * TSTEPS + t) * XDIM + c] * mk;
        else if (c == XDIM) v = mk;
        D[i] = (__bf16)v;
    }
}

// ---------------- fp32 -> bf16 weight conversion (with optional K padding) ----------------
__global__ void cvt_bf16_kernel(const float* __restrict__ src, __bf16* __restrict__ dst,
                                int rows, int srcld, int dstld, int cols) {
    int total = rows * dstld;
    for (int i = blockIdx.x * blockDim.x + threadIdx.x; i < total;
         i += gridDim.x * blockDim.x) {
        int r = i / dstld, c = i - r * dstld;
        float v = (c < cols) ? src[(size_t)r * srcld + c] : 0.f;
        dst[i] = (__bf16)v;
    }
}

// ---------------- main persistent ODE-RNN kernel ----------------
__global__ __launch_bounds__(NTHREADS)
void odernn_main(const float* __restrict__ x, const float* __restrict__ tg,
                 const int* __restrict__ mask,
                 const float* __restrict__ ode_b1, const float* __restrict__ ode_b2,
                 const float* __restrict__ b_ih,   const float* __restrict__ b_hh,
                 const float* __restrict__ out_b1, const float* __restrict__ out_b2,
                 const __bf16* __restrict__ w1b,  const __bf16* __restrict__ w2b,
                 const __bf16* __restrict__ whhb, const __bf16* __restrict__ ow1b,
                 const __bf16* __restrict__ ow2b, const __bf16* __restrict__ wihb,
                 float* __restrict__ outp, float* __restrict__ zfin) {
    extern __shared__ char smem[];
    float*  hS = (float*)smem;                                   // MT*ZDIM fp32 (64 KB)
    __bf16* P  = (__bf16*)(smem + MT * ZDIM * sizeof(float));    // MT*ZDIM bf16 (32 KB)
    __bf16* Q  = P + MT * ZDIM;                                  // MT*ZDIM bf16 (32 KB)
    __bf16* D  = Q + MT * ZDIM;                                  // MT*LDD  bf16 (6 KB)

    const int tid   = threadIdx.x;
    const int lane  = tid & 31;
    const int wave  = tid >> 5;
    const int nl    = lane & 15;
    const int half  = lane >> 4;
    const int mi    = wave & 1;      // which 16-row M tile
    const int wc    = wave >> 1;     // wave column 0..3
    const int m0    = blockIdx.x * MT;
    const int mrow0 = mi * 16;

    for (int i = tid; i < MT * ZDIM; i += NTHREADS) { hS[i] = 0.f; P[i] = (__bf16)0.f; }
    build_data(D, x, mask, m0, 0);
    __syncthreads();

    __bf16* cur = P;   // bf16 copy of current hidden state
    __bf16* alt = Q;

    for (int t = 0; t < TSTEPS; ++t) {
        float dt = (t == 0) ? 0.f : (tg[t] - tg[t - 1]);

        // -------- Phase A: alt = tanh(cur @ ode_w1^T + b1)  [MT x H] --------
        for (int g = wc; g < HDIM / 64; g += 4) {
            int nc = g * 64;   // four adjacent tiles
            f32x8 a0 = bias_acc(ode_b1[nc + nl]);
            f32x8 a1 = bias_acc(ode_b1[nc + 16 + nl]);
            f32x8 a2 = bias_acc(ode_b1[nc + 32 + nl]);
            f32x8 a3 = bias_acc(ode_b1[nc + 48 + nl]);
            wmma_tile4(cur, ZDIM, mrow0, w1b, ZDIM, nc, ZDIM, a0, a1, a2, a3, nl, half);
#pragma unroll
            for (int j = 0; j < 8; ++j) {
                int m = mrow0 + j + 8 * half;
                int i0 = m * ZDIM + nc + nl;
                alt[i0]      = (__bf16)fast_tanh(a0[j]);
                alt[i0 + 16] = (__bf16)fast_tanh(a1[j]);
                alt[i0 + 32] = (__bf16)fast_tanh(a2[j]);
                alt[i0 + 48] = (__bf16)fast_tanh(a3[j]);
            }
        }
        __syncthreads();

        // -------- Phase B: f = alt @ ode_w2^T + b2;  h += dt*f;  cur = bf16(h) --------
        for (int g = wc; g < ZDIM / 64; g += 4) {
            int nc = g * 64;
            f32x8 a0 = bias_acc(ode_b2[nc + nl]);
            f32x8 a1 = bias_acc(ode_b2[nc + 16 + nl]);
            f32x8 a2 = bias_acc(ode_b2[nc + 32 + nl]);
            f32x8 a3 = bias_acc(ode_b2[nc + 48 + nl]);
            wmma_tile4(alt, ZDIM, mrow0, w2b, HDIM, nc, HDIM, a0, a1, a2, a3, nl, half);
#pragma unroll
            for (int j = 0; j < 8; ++j) {
                int m = mrow0 + j + 8 * half;
                int i0 = m * ZDIM + nc + nl;
                float h0 = hS[i0]      + dt * a0[j];
                float h1 = hS[i0 + 16] + dt * a1[j];
                float h2 = hS[i0 + 32] + dt * a2[j];
                float h3 = hS[i0 + 48] + dt * a3[j];
                hS[i0]      = h0;  cur[i0]      = (__bf16)h0;
                hS[i0 + 16] = h1;  cur[i0 + 16] = (__bf16)h1;
                hS[i0 + 32] = h2;  cur[i0 + 32] = (__bf16)h2;
                hS[i0 + 48] = h3;  cur[i0 + 48] = (__bf16)h3;
            }
        }
        __syncthreads();

        // -------- Phase C: GRU gates; new h -> hS + alt --------
        for (int nt = wc; nt < ZDIM / 16; nt += 4) {
            int ncol = nt * 16;
            f32x8 gr  = bias_acc(b_ih[ncol + nl]            + b_hh[ncol + nl]);
            f32x8 gz  = bias_acc(b_ih[ZDIM + ncol + nl]     + b_hh[ZDIM + ncol + nl]);
            f32x8 gin = bias_acc(b_ih[2 * ZDIM + ncol + nl]);
            f32x8 ghn = bias_acc(b_hh[2 * ZDIM + ncol + nl]);
            // hidden contributions: r/z/n sections share A fragments (K = 512)
            wmma_tile3(cur, ZDIM, mrow0, whhb, ZDIM, ncol, ZDIM, gr, gz, ghn, nl, half);
            // input contributions: r/z/n sections share A fragments (K = 96, zero-padded)
            wmma_tile3(D, LDD, mrow0, wihb, LDD, ncol, LDD, gr, gz, gin, nl, half);
#pragma unroll
            for (int j = 0; j < 8; ++j) {
                int m = mrow0 + j + 8 * half;
                int idx = m * ZDIM + ncol + nl;
                float r  = fast_sigmoid(gr[j]);
                float zg = fast_sigmoid(gz[j]);
                float ng = fast_tanh(gin[j] + r * ghn[j]);
                float hv = (1.f - zg) * ng + zg * hS[idx];
                hS[idx]  = hv;
                alt[idx] = (__bf16)hv;
            }
        }
        __syncthreads();

        // -------- Phase D: cur = tanh(alt @ out_w1^T + out_b1) --------
        for (int g = wc; g < HDIM / 64; g += 4) {
            int nc = g * 64;
            f32x8 a0 = bias_acc(out_b1[nc + nl]);
            f32x8 a1 = bias_acc(out_b1[nc + 16 + nl]);
            f32x8 a2 = bias_acc(out_b1[nc + 32 + nl]);
            f32x8 a3 = bias_acc(out_b1[nc + 48 + nl]);
            wmma_tile4(alt, ZDIM, mrow0, ow1b, ZDIM, nc, ZDIM, a0, a1, a2, a3, nl, half);
#pragma unroll
            for (int j = 0; j < 8; ++j) {
                int m = mrow0 + j + 8 * half;
                int i0 = m * ZDIM + nc + nl;
                cur[i0]      = (__bf16)fast_tanh(a0[j]);
                cur[i0 + 16] = (__bf16)fast_tanh(a1[j]);
                cur[i0 + 32] = (__bf16)fast_tanh(a2[j]);
                cur[i0 + 48] = (__bf16)fast_tanh(a3[j]);
            }
        }
        __syncthreads();

        // -------- Phase E: out[:, t, :] = cur @ out_w2^T + out_b2  [MT x 64] --------
        {
            int nt = wc;   // 4 tiles x 2 M-tiles = 8 wave tasks
            f32x8 acc = bias_acc(out_b2[nt * 16 + nl]);
            acc = wmma_tile1(cur, ZDIM, mrow0, ow2b, HDIM, nt * 16, HDIM, acc, nl, half);
#pragma unroll
            for (int j = 0; j < 8; ++j) {
                int m = mrow0 + j + 8 * half;
                int b = m0 + m;
                outp[((size_t)b * TSTEPS + t) * XDIM + nt * 16 + nl] = acc[j];
            }
        }

        // -------- Phase F: stage GRU input for t+1 --------
        if (t + 1 < TSTEPS) build_data(D, x, mask, m0, t + 1);
        __syncthreads();

        __bf16* tmp = cur; cur = alt; alt = tmp;   // new state lives in alt
    }

    // z_final from the fp32 master state
    for (int i = tid; i < MT * ZDIM; i += NTHREADS) {
        int r = i / ZDIM, c = i - r * ZDIM;
        zfin[(size_t)(m0 + r) * ZDIM + c] = hS[i];
    }
}

// ---------------- host launcher ----------------
extern "C" void kernel_launch(void* const* d_in, const int* in_sizes, int n_in,
                              void* d_out, int out_size, void* d_ws, size_t ws_size,
                              hipStream_t stream) {
    (void)in_sizes; (void)n_in; (void)out_size; (void)ws_size;

    const float* x      = (const float*)d_in[0];
    const float* tg     = (const float*)d_in[1];
    const int*   mask   = (const int*)  d_in[2];
    const float* ode_w1 = (const float*)d_in[3];
    const float* ode_b1 = (const float*)d_in[4];
    const float* ode_w2 = (const float*)d_in[5];
    const float* ode_b2 = (const float*)d_in[6];
    const float* w_ih   = (const float*)d_in[7];
    const float* w_hh   = (const float*)d_in[8];
    const float* b_ih   = (const float*)d_in[9];
    const float* b_hh   = (const float*)d_in[10];
    const float* out_w1 = (const float*)d_in[11];
    const float* out_b1 = (const float*)d_in[12];
    const float* out_w2 = (const float*)d_in[13];
    const float* out_b2 = (const float*)d_in[14];

    // bf16 weight copies in workspace (torch [out][in] layout == WMMA B-fragment layout)
    __bf16* w1b  = (__bf16*)d_ws;
    __bf16* w2b  = w1b  + 512 * 512;
    __bf16* whhb = w2b  + 512 * 512;
    __bf16* ow1b = whhb + (size_t)1536 * 512;
    __bf16* ow2b = ow1b + 512 * 512;
    __bf16* wihb = ow2b + 64 * 512;     // padded [1536 x 96]

    cvt_bf16_kernel<<<256, 256, 0, stream>>>(ode_w1, w1b, 512,  512, 512, 512);
    cvt_bf16_kernel<<<256, 256, 0, stream>>>(ode_w2, w2b, 512,  512, 512, 512);
    cvt_bf16_kernel<<<256, 256, 0, stream>>>(w_hh,  whhb, 1536, 512, 512, 512);
    cvt_bf16_kernel<<<256, 256, 0, stream>>>(out_w1, ow1b, 512, 512, 512, 512);
    cvt_bf16_kernel<<<256, 256, 0, stream>>>(out_w2, ow2b, 64,  512, 512, 512);
    cvt_bf16_kernel<<<256, 256, 0, stream>>>(w_ih,  wihb, 1536, 65,  96,  65);

    float* outp = (float*)d_out;
    float* zfin = outp + (size_t)BATCH * TSTEPS * XDIM;

    constexpr size_t shmem = MT * ZDIM * sizeof(float)        // hS
                           + 2 * MT * ZDIM * sizeof(__bf16)   // P, Q
                           + MT * LDD * sizeof(__bf16);       // D   == 137216 B
    (void)hipFuncSetAttribute((const void*)odernn_main,
                              hipFuncAttributeMaxDynamicSharedMemorySize, (int)shmem);

    odernn_main<<<BATCH / MT, NTHREADS, shmem, stream>>>(
        x, tg, mask, ode_b1, ode_b2, b_ih, b_hh, out_b1, out_b2,
        w1b, w2b, whhb, ow1b, ow2b, wihb, outp, zfin);
}
